// NystromSelfAttention_35167192220034
// MI455X (gfx1250) — matrix operand order repaired and checked by reading, not compiled, not run on hardware
//
#include <hip/hip_runtime.h>

typedef __attribute__((ext_vector_type(16))) __bf16 v16bf;
typedef __attribute__((ext_vector_type(8)))  float  v8f;
typedef __attribute__((ext_vector_type(2)))  float  v2f;
typedef __attribute__((ext_vector_type(4)))  unsigned int u32x4;
typedef __attribute__((ext_vector_type(8)))  int i32x8;
typedef __attribute__((ext_vector_type(4)))  int i32x4;

static constexpr int Bn  = 4;
static constexpr int Sn  = 4096;
static constexpr int En  = 1024;
static constexpr int Dn  = 512;   // DK == DV
static constexpr int Mn  = 64;

__device__ __forceinline__ v8f v8f_zero() {
  v8f z = {0.f, 0.f, 0.f, 0.f, 0.f, 0.f, 0.f, 0.f};
  return z;
}

__device__ __forceinline__ unsigned short f32_to_bf16(float f) {
  unsigned int u = __float_as_uint(f);
  u += 0x7FFFu + ((u >> 16) & 1u);
  return (unsigned short)(u >> 16);
}
__device__ __forceinline__ float bf16_to_f32(unsigned short h) {
  return __uint_as_float(((unsigned int)h) << 16);
}

__device__ __forceinline__ v8f wmma_bf16(v16bf a, v16bf b, v8f c) {
  return __builtin_amdgcn_wmma_f32_16x16x32_bf16(false, a, false, b, (short)0, c, false, false);
}

// ---------------- Tensor Data Mover: 2D bf16 tile, global -> LDS ------------------
// tile_k elements per row (dim0), rows (dim1), row stride in elements.
// pad_int_code / pad_amt_code per D# encoding (interval DWORDs = 2<<code, amount = (code+1) DWORDs).
__device__ __forceinline__ void tdm_load_2d_bf16(const unsigned short* gptr, unsigned lds_off,
                                                 unsigned tile_k, unsigned rows,
                                                 unsigned stride_elems,
                                                 int pad_int_code, int pad_amt_code) {
  unsigned long long ga = (unsigned long long)(size_t)gptr;
  u32x4 g0;
  g0[0] = 1u;                                             // count=1, user descriptor
  g0[1] = lds_off;                                        // lds_addr (bytes)
  g0[2] = (unsigned)(ga & 0xFFFFFFFFu);                   // global_addr[31:0]
  g0[3] = (unsigned)((ga >> 32) & 0x1FFFFFFu) | 0x80000000u; // global_addr[56:32] | type=2
  unsigned pad_en = (pad_amt_code >= 0) ? 1u : 0u;
  i32x8 g1;
  g1[0] = (int)((1u << 16)                                // data_size = 2 bytes
         | (pad_en << 20)
         | ((pad_en ? (unsigned)pad_int_code : 0u) << 22)
         | ((pad_en ? (unsigned)pad_amt_code : 0u) << 25));
  g1[1] = (int)((tile_k & 0xFFFFu) << 16);                // tensor_dim0[15:0] @bits 63:48
  g1[2] = (int)((tile_k >> 16) | ((rows & 0xFFFFu) << 16)); // tensor_dim0 hi | tensor_dim1 lo
  g1[3] = (int)((rows >> 16) | ((tile_k & 0xFFFFu) << 16)); // tensor_dim1 hi | tile_dim0
  g1[4] = (int)(rows & 0xFFFFu);                          // tile_dim1 (tile_dim2 = 0)
  g1[5] = (int)stride_elems;                              // tensor_dim0_stride[31:0]
  g1[6] = 0;                                              // stride0 hi16 | dim1_stride lo16
  g1[7] = 0;
  i32x4 z4 = {0, 0, 0, 0};
  i32x8 z8 = {0, 0, 0, 0, 0, 0, 0, 0};
  __builtin_amdgcn_tensor_load_to_lds(g0, g1, z4, z4, z8, 0);
}

__device__ __forceinline__ unsigned lds_off_of(const void* p) {
  return (unsigned)(size_t)p;   // low 32 bits of the LDS aperture address = LDS offset
}

union FragU { uint4 u[2]; v16bf v; };

// A fragment 16x32 bf16: lanes<16 hold K {0..7,16..23}, lanes>=16 {8..15,24..31}
__device__ __forceinline__ v16bf frag_a(const unsigned short* lds, int stride,
                                        int row0, int koff, int lane) {
  int row = row0 + (lane & 15);
  const unsigned short* p = lds + row * stride + koff + ((lane & 16) >> 1);
  FragU f;
  f.u[0] = *(const uint4*)(p);
  f.u[1] = *(const uint4*)(p + 16);
  return f.v;
}

// B fragment 32x16 bf16 from LDS staged as Bt[n][k]
__device__ __forceinline__ v16bf frag_b(const unsigned short* lds, int stride,
                                        int col0, int koff, int lane) {
  int col = col0 + (lane & 15);
  const unsigned short* p = lds + col * stride + koff + ((lane & 16) >> 1);
  FragU f;
  f.u[0] = *(const uint4*)(p);
  f.u[1] = *(const uint4*)(p + 16);
  return f.v;
}

// f32 global tile -> bf16 LDS (row major), for fp32 sources only
__device__ __forceinline__ void load_ab(const float* __restrict__ g, int ld,
                                        unsigned short* lds, int stride,
                                        int rows, int kk, int tid, int nthr) {
  int total = rows * kk;
  for (int i = tid * 4; i < total; i += nthr * 4) {
    int r = i / kk, c = i - r * kk;
    float4 f = *(const float4*)(g + (size_t)r * ld + c);
    unsigned short* d = lds + r * stride + c;
    d[0] = f32_to_bf16(f.x); d[1] = f32_to_bf16(f.y);
    d[2] = f32_to_bf16(f.z); d[3] = f32_to_bf16(f.w);
  }
}

// f32 global tile -> transposed bf16 LDS Bt[cols][stride]
__device__ __forceinline__ void load_bT(const float* __restrict__ g, int ld,
                                        unsigned short* lds, int stride,
                                        int kk, int cols, int tid, int nthr) {
  int total = kk * cols;
  for (int i = tid * 4; i < total; i += nthr * 4) {
    int r = i / cols, c = i - r * cols;
    float4 f = *(const float4*)(g + (size_t)r * ld + c);
    lds[(c + 0) * stride + r] = f32_to_bf16(f.x);
    lds[(c + 1) * stride + r] = f32_to_bf16(f.y);
    lds[(c + 2) * stride + r] = f32_to_bf16(f.z);
    lds[(c + 3) * stride + r] = f32_to_bf16(f.w);
  }
}

__device__ __forceinline__ void store_tile_g(float* __restrict__ g, int ld,
                                             int row0, int col0, v8f c, int lane) {
  int col = col0 + (lane & 15);
  int r0  = row0 + ((lane & 16) >> 1);
#pragma unroll
  for (int r = 0; r < 8; ++r) g[(size_t)(r0 + r) * ld + col] = c[r];
}

__device__ __forceinline__ void store_tile_bf16(unsigned short* __restrict__ g, int ld,
                                                int row0, int col0, v8f c, int lane) {
  int col = col0 + (lane & 15);
  int r0  = row0 + ((lane & 16) >> 1);
#pragma unroll
  for (int r = 0; r < 8; ++r) g[(size_t)(r0 + r) * ld + col] = f32_to_bf16(c[r]);
}

__device__ __forceinline__ void store_tile_lds(float* s, int ld,
                                               int row0, int col0, v8f c, int lane) {
  int col = col0 + (lane & 15);
  int r0  = row0 + ((lane & 16) >> 1);
#pragma unroll
  for (int r = 0; r < 8; ++r) s[(r0 + r) * ld + col] = c[r];
}

// ---------------- projection: out_bf16 = (X @ W + bias) * scale -------------------
// transp==false: out[16384][512]; transp==true: out stored as [B][512][4096] (for V)
__global__ __launch_bounds__(256) void proj_kernel(const float* __restrict__ X,
                                                   const float* __restrict__ W,
                                                   const float* __restrict__ bias,
                                                   unsigned short* __restrict__ out,
                                                   float scale, int transp) {
  __shared__ alignas(16) unsigned short Al[128 * 40];
  __shared__ alignas(16) unsigned short Bl[128 * 40];
  const int tid = threadIdx.x, lane = tid & 31, wave = tid >> 5;
  const int rb = blockIdx.x * 128, cb = blockIdx.y * 128;
  const int wm = wave >> 1, wn = wave & 1;   // 4x2 waves, 32x64 per wave
  v8f acc[2][4];
#pragma unroll
  for (int i = 0; i < 2; ++i)
#pragma unroll
    for (int j = 0; j < 4; ++j) acc[i][j] = v8f_zero();
  for (int k0 = 0; k0 < En; k0 += 32) {
    load_ab(X + (size_t)rb * En + k0, En, Al, 40, 128, 32, tid, 256);
    load_bT(W + (size_t)k0 * Dn + cb, Dn, Bl, 40, 32, 128, tid, 256);
    __syncthreads();
    v16bf fa[2], fb[4];
#pragma unroll
    for (int i = 0; i < 2; ++i) fa[i] = frag_a(Al, 40, wm * 32 + i * 16, 0, lane);
#pragma unroll
    for (int j = 0; j < 4; ++j) fb[j] = frag_b(Bl, 40, wn * 64 + j * 16, 0, lane);
#pragma unroll
    for (int i = 0; i < 2; ++i)
#pragma unroll
      for (int j = 0; j < 4; ++j) acc[i][j] = wmma_bf16(fa[i], fb[j], acc[i][j]);
    __syncthreads();
  }
#pragma unroll
  for (int i = 0; i < 2; ++i)
#pragma unroll
    for (int j = 0; j < 4; ++j) {
      int col = cb + wn * 64 + j * 16 + (lane & 15);
      int row0 = rb + wm * 32 + i * 16 + ((lane & 16) >> 1);
      float bv = bias[col];
      v8f c = acc[i][j];
      if (!transp) {
#pragma unroll
        for (int r = 0; r < 8; ++r)
          out[(size_t)(row0 + r) * Dn + col] = f32_to_bf16((c[r] + bv) * scale);
      } else {
#pragma unroll
        for (int r = 0; r < 8; ++r) {
          int rr = row0 + r, bb = rr >> 12, sr = rr & 4095;
          out[((size_t)bb * Dn + col) * Sn + sr] = f32_to_bf16((c[r] + bv) * scale);
        }
      }
    }
}

// ---------------- landmark pooling (bf16 in / bf16 out) ---------------------------
__global__ __launch_bounds__(256) void pool_kernel(const unsigned short* __restrict__ src,
                                                   unsigned short* __restrict__ dst) {
  int b = blockIdx.x >> 6, m = blockIdx.x & 63, tid = threadIdx.x;
  const unsigned short* p = src + ((size_t)b * Sn + (size_t)m * 64) * Dn;
  unsigned short* q = dst + ((size_t)b * Mn + m) * Dn;
  for (int c = tid; c < Dn; c += 256) {
    float s = 0.f;
#pragma unroll 8
    for (int r = 0; r < 64; ++r) s += bf16_to_f32(p[(size_t)r * Dn + c]);
    q[c] = f32_to_bf16(s * (1.0f / 64.0f));
  }
}

// ---------------- kernel_1 = softmax(q @ k_lm^T)  (TDM double-buffered) -----------
__global__ __launch_bounds__(256) void k1_kernel(const unsigned short* __restrict__ q,
                                                 const unsigned short* __restrict__ klm,
                                                 unsigned short* __restrict__ K1) {
  __shared__ alignas(16) unsigned short Al[2][128 * 40];
  __shared__ alignas(16) unsigned short Bl[2][64 * 40];
  __shared__ float Sm[128 * 72];
  const int tid = threadIdx.x, lane = tid & 31, wave = tid >> 5;
  const int b = blockIdx.y, rb = blockIdx.x * 128;
  const int wm = wave >> 1, wn = wave & 1;
  const unsigned short* qb = q + ((size_t)b * Sn + rb) * Dn;
  const unsigned short* kb = klm + (size_t)b * Mn * Dn;
  v8f acc[2][2];
#pragma unroll
  for (int i = 0; i < 2; ++i)
#pragma unroll
    for (int j = 0; j < 2; ++j) acc[i][j] = v8f_zero();
  const int nk = Dn / 32;
  if (wave == 0 && lane == 0) {
    tdm_load_2d_bf16(qb, lds_off_of(&Al[0][0]), 32, 128, Dn, 3, 3);
    tdm_load_2d_bf16(kb, lds_off_of(&Bl[0][0]), 32, 64, Dn, 3, 3);
  }
  for (int it = 0; it < nk; ++it) {
    if (wave == 0 && lane == 0) {
      if (it + 1 < nk) {
        tdm_load_2d_bf16(qb + (it + 1) * 32, lds_off_of(&Al[(it + 1) & 1][0]), 32, 128, Dn, 3, 3);
        tdm_load_2d_bf16(kb + (it + 1) * 32, lds_off_of(&Bl[(it + 1) & 1][0]), 32, 64, Dn, 3, 3);
        __builtin_amdgcn_s_wait_tensorcnt(2);
      } else {
        __builtin_amdgcn_s_wait_tensorcnt(0);
      }
    }
    __syncthreads();
    const unsigned short* Ac = &Al[it & 1][0];
    const unsigned short* Bc = &Bl[it & 1][0];
    v16bf fa[2], fb[2];
#pragma unroll
    for (int i = 0; i < 2; ++i) fa[i] = frag_a(Ac, 40, wm * 32 + i * 16, 0, lane);
#pragma unroll
    for (int j = 0; j < 2; ++j) fb[j] = frag_b(Bc, 40, wn * 32 + j * 16, 0, lane);
#pragma unroll
    for (int i = 0; i < 2; ++i)
#pragma unroll
      for (int j = 0; j < 2; ++j) acc[i][j] = wmma_bf16(fa[i], fb[j], acc[i][j]);
    __syncthreads();
  }
#pragma unroll
  for (int i = 0; i < 2; ++i)
#pragma unroll
    for (int j = 0; j < 2; ++j)
      store_tile_lds(Sm, 72, wm * 32 + i * 16, wn * 32 + j * 16, acc[i][j], lane);
  __syncthreads();
  if (tid < 128) {
    float mx = -1e30f;
    for (int m = 0; m < 64; ++m) mx = fmaxf(mx, Sm[tid * 72 + m]);
    float s = 0.f;
    for (int m = 0; m < 64; ++m) {
      float e = __expf(Sm[tid * 72 + m] - mx);
      Sm[tid * 72 + m] = e;
      s += e;
    }
    float inv = 1.f / s;
    unsigned short* o = K1 + ((size_t)b * Sn + rb + tid) * Mn;
    for (int m = 0; m < 64; ++m) o[m] = f32_to_bf16(Sm[tid * 72 + m] * inv);
  }
}

// ---------------- 64x64 f32 matmul on LDS via V_WMMA_F32_16X16X4_F32 --------------
#define MST 66
__device__ void mm64(const float* A, const float* Bm, float* D, int tid) {
  const int lane = tid & 31, wave = tid >> 5;
  const int l15 = lane & 15, g2 = (lane & 16) >> 3;
#pragma unroll
  for (int t = 0; t < 2; ++t) {
    int tt = wave * 2 + t;
    int tm = (tt >> 2) << 4, tn = (tt & 3) << 4;
    v8f c = v8f_zero();
#pragma unroll
    for (int k0 = 0; k0 < 64; k0 += 4) {
      int ka = k0 + g2;
      v2f a, b;
      a.x = A[(tm + l15) * MST + ka];
      a.y = A[(tm + l15) * MST + ka + 1];
      b.x = Bm[ka * MST + tn + l15];
      b.y = Bm[(ka + 1) * MST + tn + l15];
      c = __builtin_amdgcn_wmma_f32_16x16x4_f32(false, a, false, b, (short)0, c, false, false);
    }
    int r0 = tm + ((lane & 16) >> 1);
#pragma unroll
    for (int r = 0; r < 8; ++r) D[(r0 + r) * MST + tn + l15] = c[r];
  }
}

// ---------------- kernel_2 = softmax(q_lm @ k_lm^T); pinv via Newton-Schulz -------
__global__ __launch_bounds__(256) void k2pinv_kernel(const unsigned short* __restrict__ qlm,
                                                     const unsigned short* __restrict__ klm,
                                                     float* __restrict__ k2inv) {
  __shared__ float Am[64 * MST];
  __shared__ float Zm[64 * MST];
  __shared__ alignas(16) float Pm[64 * MST];   // phase 1 reuses as bf16 staging
  __shared__ float Tm[64 * MST];
  __shared__ float Um[64 * MST];
  __shared__ float red[64];
  __shared__ float scal;
  const int tid = threadIdx.x, lane = tid & 31, wave = tid >> 5;
  const int b = blockIdx.x;
  unsigned short* Alb = (unsigned short*)Pm;
  unsigned short* Blb = Alb + 64 * 40;
  const unsigned short* qb = qlm + (size_t)b * Mn * Dn;
  const unsigned short* kb = klm + (size_t)b * Mn * Dn;
  v8f acc[2];
  acc[0] = v8f_zero(); acc[1] = v8f_zero();
  for (int k0 = 0; k0 < Dn; k0 += 32) {
    if (wave == 0 && lane == 0) {
      tdm_load_2d_bf16(qb + k0, lds_off_of(Alb), 32, 64, Dn, 3, 3);
      tdm_load_2d_bf16(kb + k0, lds_off_of(Blb), 32, 64, Dn, 3, 3);
      __builtin_amdgcn_s_wait_tensorcnt(0);
    }
    __syncthreads();
#pragma unroll
    for (int t = 0; t < 2; ++t) {
      int tt = wave * 2 + t, tm = (tt >> 2) << 4, tn = (tt & 3) << 4;
      v16bf fa = frag_a(Alb, 40, tm, 0, lane);
      v16bf fb = frag_b(Blb, 40, tn, 0, lane);
      acc[t] = wmma_bf16(fa, fb, acc[t]);
    }
    __syncthreads();
  }
#pragma unroll
  for (int t = 0; t < 2; ++t) {
    int tt = wave * 2 + t, tm = (tt >> 2) << 4, tn = (tt & 3) << 4;
    store_tile_lds(Am, MST, tm, tn, acc[t], lane);
  }
  __syncthreads();
  if (tid < 64) {
    float mx = -1e30f;
    for (int j = 0; j < 64; ++j) mx = fmaxf(mx, Am[tid * MST + j]);
    float s = 0.f;
    for (int j = 0; j < 64; ++j) {
      float e = __expf(Am[tid * MST + j] - mx);
      Am[tid * MST + j] = e;
      s += e;
    }
    float inv = 1.f / s;
    for (int j = 0; j < 64; ++j) Am[tid * MST + j] *= inv;
  }
  __syncthreads();
  if (tid < 64) {
    float s = 0.f;
    for (int i = 0; i < 64; ++i) s += fabsf(Am[i * MST + tid]);
    red[tid] = s;
  }
  __syncthreads();
  if (tid == 0) {
    float cm = 0.f;
    for (int i = 0; i < 64; ++i) cm = fmaxf(cm, red[i]);
    scal = 1.0f / cm;
  }
  __syncthreads();
  for (int i = tid; i < 4096; i += 256) {
    int r = i >> 6, c = i & 63;
    Zm[r * MST + c] = Am[c * MST + r] * scal;
  }
  __syncthreads();
  for (int it = 0; it < 8; ++it) {
    mm64(Am, Zm, Pm, tid); __syncthreads();
    for (int i = tid; i < 4096; i += 256) {
      int r = i >> 6, c = i & 63;
      Tm[r * MST + c] = (r == c ? 7.f : 0.f) - Pm[r * MST + c];
    }
    __syncthreads();
    mm64(Pm, Tm, Um, tid); __syncthreads();
    for (int i = tid; i < 4096; i += 256) {
      int r = i >> 6, c = i & 63;
      Um[r * MST + c] = (r == c ? 15.f : 0.f) - Um[r * MST + c];
    }
    __syncthreads();
    mm64(Pm, Um, Tm, tid); __syncthreads();
    for (int i = tid; i < 4096; i += 256) {
      int r = i >> 6, c = i & 63;
      Tm[r * MST + c] = (r == c ? 13.f : 0.f) - Tm[r * MST + c];
    }
    __syncthreads();
    mm64(Zm, Tm, Um, tid); __syncthreads();
    for (int i = tid; i < 4096; i += 256) {
      int r = i >> 6, c = i & 63;
      Zm[r * MST + c] = 0.25f * Um[r * MST + c];
    }
    __syncthreads();
  }
  float* o = k2inv + (size_t)b * 4096;
  for (int i = tid; i < 4096; i += 256) {
    int r = i >> 6, c = i & 63;
    o[i] = Zm[r * MST + c];
  }
}

// ---------------- scores3 = q_lm @ k^T (TDM double-buffered) ----------------------
__global__ __launch_bounds__(256) void k3_kernel(const unsigned short* __restrict__ qlm,
                                                 const unsigned short* __restrict__ k,
                                                 float* __restrict__ S3) {
  __shared__ alignas(16) unsigned short Al[2][64 * 40];
  __shared__ alignas(16) unsigned short Bl[2][128 * 40];
  const int tid = threadIdx.x, lane = tid & 31, wave = tid >> 5;
  const int b = blockIdx.y, cb = blockIdx.x * 128;
  const int wm = wave >> 2, wn = wave & 3;
  const unsigned short* qb = qlm + (size_t)b * Mn * Dn;
  const unsigned short* kb = k + ((size_t)b * Sn + cb) * Dn;
  v8f acc[2][2];
#pragma unroll
  for (int i = 0; i < 2; ++i)
#pragma unroll
    for (int j = 0; j < 2; ++j) acc[i][j] = v8f_zero();
  const int nk = Dn / 32;
  if (wave == 0 && lane == 0) {
    tdm_load_2d_bf16(qb, lds_off_of(&Al[0][0]), 32, 64, Dn, 3, 3);
    tdm_load_2d_bf16(kb, lds_off_of(&Bl[0][0]), 32, 128, Dn, 3, 3);
  }
  for (int it = 0; it < nk; ++it) {
    if (wave == 0 && lane == 0) {
      if (it + 1 < nk) {
        tdm_load_2d_bf16(qb + (it + 1) * 32, lds_off_of(&Al[(it + 1) & 1][0]), 32, 64, Dn, 3, 3);
        tdm_load_2d_bf16(kb + (it + 1) * 32, lds_off_of(&Bl[(it + 1) & 1][0]), 32, 128, Dn, 3, 3);
        __builtin_amdgcn_s_wait_tensorcnt(2);
      } else {
        __builtin_amdgcn_s_wait_tensorcnt(0);
      }
    }
    __syncthreads();
    const unsigned short* Ac = &Al[it & 1][0];
    const unsigned short* Bc = &Bl[it & 1][0];
    v16bf fa[2], fb[2];
#pragma unroll
    for (int i = 0; i < 2; ++i) fa[i] = frag_a(Ac, 40, wm * 32 + i * 16, 0, lane);
#pragma unroll
    for (int j = 0; j < 2; ++j) fb[j] = frag_b(Bc, 40, wn * 32 + j * 16, 0, lane);
#pragma unroll
    for (int i = 0; i < 2; ++i)
#pragma unroll
      for (int j = 0; j < 2; ++j) acc[i][j] = wmma_bf16(fa[i], fb[j], acc[i][j]);
    __syncthreads();
  }
  float* o = S3 + (size_t)b * Mn * Sn;
#pragma unroll
  for (int i = 0; i < 2; ++i)
#pragma unroll
    for (int j = 0; j < 2; ++j)
      store_tile_g(o, Sn, wm * 32 + i * 16, cb + wn * 32 + j * 16, acc[i][j], lane);
}

// ---------------- row softmax over 4096; writes f32 in place + bf16 copy ----------
__global__ __launch_bounds__(256) void softmax_row_kernel(float* __restrict__ S3,
                                                          unsigned short* __restrict__ S3b) {
  __shared__ float red[256];
  const int tid = threadIdx.x;
  float* p = S3 + (size_t)blockIdx.x * Sn;
  unsigned short* pb = S3b + (size_t)blockIdx.x * Sn;
  float mx = -1e30f;
  for (int i = tid; i < Sn; i += 256) mx = fmaxf(mx, p[i]);
  red[tid] = mx; __syncthreads();
  for (int o = 128; o > 0; o >>= 1) {
    if (tid < o) red[tid] = fmaxf(red[tid], red[tid + o]);
    __syncthreads();
  }
  mx = red[0]; __syncthreads();
  float s = 0.f;
  for (int i = tid; i < Sn; i += 256) {
    float e = __expf(p[i] - mx);
    p[i] = e; s += e;
  }
  red[tid] = s; __syncthreads();
  for (int o = 128; o > 0; o >>= 1) {
    if (tid < o) red[tid] += red[tid + o];
    __syncthreads();
  }
  float inv = 1.0f / red[0];
  for (int i = tid; i < Sn; i += 256) pb[i] = f32_to_bf16(p[i] * inv);
}

// ---------------- k3v^T = (kernel_3 @ v)^T stored [B,512,64] ----------------------
// A = S3b [64 x 4096] bf16, B = vT [512 x 4096] bf16 (Bt[n][k] = vT rows directly)
__global__ __launch_bounds__(256) void k3v_kernel(const unsigned short* __restrict__ S3b,
                                                  const unsigned short* __restrict__ vT,
                                                  unsigned short* __restrict__ k3vT) {
  __shared__ alignas(16) unsigned short Al[2][64 * 40];
  __shared__ alignas(16) unsigned short Bl[2][128 * 40];
  const int tid = threadIdx.x, lane = tid & 31, wave = tid >> 5;
  const int b = blockIdx.y, cb = blockIdx.x * 128;
  const int wm = wave >> 2, wn = wave & 3;
  const unsigned short* ab = S3b + (size_t)b * Mn * Sn;
  const unsigned short* vb = vT + (size_t)b * Dn * Sn + (size_t)cb * Sn;
  v8f acc[2][2];
#pragma unroll
  for (int i = 0; i < 2; ++i)
#pragma unroll
    for (int j = 0; j < 2; ++j) acc[i][j] = v8f_zero();
  const int nk = Sn / 32;
  if (wave == 0 && lane == 0) {
    tdm_load_2d_bf16(ab, lds_off_of(&Al[0][0]), 32, 64, Sn, 3, 3);
    tdm_load_2d_bf16(vb, lds_off_of(&Bl[0][0]), 32, 128, Sn, 3, 3);
  }
  for (int it = 0; it < nk; ++it) {
    if (wave == 0 && lane == 0) {
      if (it + 1 < nk) {
        tdm_load_2d_bf16(ab + (it + 1) * 32, lds_off_of(&Al[(it + 1) & 1][0]), 32, 64, Sn, 3, 3);
        tdm_load_2d_bf16(vb + (it + 1) * 32, lds_off_of(&Bl[(it + 1) & 1][0]), 32, 128, Sn, 3, 3);
        __builtin_amdgcn_s_wait_tensorcnt(2);
      } else {
        __builtin_amdgcn_s_wait_tensorcnt(0);
      }
    }
    __syncthreads();
    const unsigned short* Ac = &Al[it & 1][0];
    const unsigned short* Bc = &Bl[it & 1][0];
    v16bf fa[2], fb[2];
#pragma unroll
    for (int i = 0; i < 2; ++i) fa[i] = frag_a(Ac, 40, wm * 32 + i * 16, 0, lane);
#pragma unroll
    for (int j = 0; j < 2; ++j) fb[j] = frag_b(Bc, 40, wn * 32 + j * 16, 0, lane);
#pragma unroll
    for (int i = 0; i < 2; ++i)
#pragma unroll
      for (int j = 0; j < 2; ++j) acc[i][j] = wmma_bf16(fa[i], fb[j], acc[i][j]);
    __syncthreads();
  }
  // store transposed: k3vT[b][d][m] with d = cb+col, m = row
  unsigned short* o = k3vT + (size_t)b * Dn * Mn;
#pragma unroll
  for (int i = 0; i < 2; ++i)
#pragma unroll
    for (int j = 0; j < 2; ++j) {
      int col = cb + wn * 32 + j * 16 + (lane & 15);
      int r0  = wm * 32 + i * 16 + ((lane & 16) >> 1);
      v8f c = acc[i][j];
#pragma unroll
      for (int r = 0; r < 8; ++r) o[(size_t)col * Mn + r0 + r] = f32_to_bf16(c[r]);
    }
}

// ---------------- tmp = kernel_1 @ k2inv (bf16 out) -------------------------------
__global__ __launch_bounds__(256) void tmp_kernel(const unsigned short* __restrict__ K1,
                                                  const float* __restrict__ k2inv,
                                                  unsigned short* __restrict__ T) {
  __shared__ alignas(16) unsigned short Al[128 * 72];
  __shared__ alignas(16) unsigned short Bl[64 * 72];
  const int tid = threadIdx.x, lane = tid & 31, wave = tid >> 5;
  const int b = blockIdx.y, rb = blockIdx.x * 128;
  const int wm = wave >> 1, wn = wave & 1;
  if (wave == 0 && lane == 0)
    tdm_load_2d_bf16(K1 + ((size_t)b * Sn + rb) * Mn, lds_off_of(Al), 64, 128, Mn, 4, 3);
  load_bT(k2inv + (size_t)b * 4096, 64, Bl, 72, 64, 64, tid, 256);
  if (wave == 0 && lane == 0) __builtin_amdgcn_s_wait_tensorcnt(0);
  __syncthreads();
  v8f acc[2][2];
#pragma unroll
  for (int i = 0; i < 2; ++i)
#pragma unroll
    for (int j = 0; j < 2; ++j) acc[i][j] = v8f_zero();
#pragma unroll
  for (int ko = 0; ko < 64; ko += 32) {
    v16bf fa[2], fb[2];
#pragma unroll
    for (int i = 0; i < 2; ++i) fa[i] = frag_a(Al, 72, wm * 32 + i * 16, ko, lane);
#pragma unroll
    for (int j = 0; j < 2; ++j) fb[j] = frag_b(Bl, 72, wn * 32 + j * 16, ko, lane);
#pragma unroll
    for (int i = 0; i < 2; ++i)
#pragma unroll
      for (int j = 0; j < 2; ++j) acc[i][j] = wmma_bf16(fa[i], fb[j], acc[i][j]);
  }
  unsigned short* o = T + ((size_t)b * Sn + rb) * Mn;
#pragma unroll
  for (int i = 0; i < 2; ++i)
#pragma unroll
    for (int j = 0; j < 2; ++j)
      store_tile_bf16(o, Mn, wm * 32 + i * 16, wn * 32 + j * 16, acc[i][j], lane);
}

// ---------------- out = tmp @ k3v  (f32 out; B from k3vT rows) --------------------
__global__ __launch_bounds__(256) void out_kernel(const unsigned short* __restrict__ T,
                                                  const unsigned short* __restrict__ k3vT,
                                                  float* __restrict__ O) {
  __shared__ alignas(16) unsigned short Al[128 * 72];
  __shared__ alignas(16) unsigned short Bl[128 * 72];
  const int tid = threadIdx.x, lane = tid & 31, wave = tid >> 5;
  const int b = blockIdx.z, rb = blockIdx.x * 128, cb = blockIdx.y * 128;
  const int wm = wave >> 1, wn = wave & 1;   // 32x64 per wave
  if (wave == 0 && lane == 0) {
    tdm_load_2d_bf16(T + ((size_t)b * Sn + rb) * Mn, lds_off_of(Al), 64, 128, Mn, 4, 3);
    tdm_load_2d_bf16(k3vT + ((size_t)b * Dn + cb) * Mn, lds_off_of(Bl), 64, 128, Mn, 4, 3);
    __builtin_amdgcn_s_wait_tensorcnt(0);
  }
  __syncthreads();
  v8f acc[2][4];
#pragma unroll
  for (int i = 0; i < 2; ++i)
#pragma unroll
    for (int j = 0; j < 4; ++j) acc[i][j] = v8f_zero();
#pragma unroll
  for (int ko = 0; ko < 64; ko += 32) {
    v16bf fa[2], fb[4];
#pragma unroll
    for (int i = 0; i < 2; ++i) fa[i] = frag_a(Al, 72, wm * 32 + i * 16, ko, lane);
#pragma unroll
    for (int j = 0; j < 4; ++j) fb[j] = frag_b(Bl, 72, wn * 64 + j * 16, ko, lane);
#pragma unroll
    for (int i = 0; i < 2; ++i)
#pragma unroll
      for (int j = 0; j < 4; ++j) acc[i][j] = wmma_bf16(fa[i], fb[j], acc[i][j]);
  }
  float* o = O + (size_t)b * Sn * Dn;
#pragma unroll
  for (int i = 0; i < 2; ++i)
#pragma unroll
    for (int j = 0; j < 4; ++j)
      store_tile_g(o, Dn, rb + wm * 32 + i * 16, cb + wn * 64 + j * 16, acc[i][j], lane);
}

extern "C" void kernel_launch(void* const* d_in, const int* in_sizes, int n_in,
                              void* d_out, int out_size, void* d_ws, size_t ws_size,
                              hipStream_t stream) {
  (void)in_sizes; (void)n_in; (void)out_size; (void)ws_size;
  const float* query = (const float*)d_in[0];
  const float* key   = (const float*)d_in[1];
  const float* value = (const float*)d_in[2];
  const float* Wq = (const float*)d_in[3];
  const float* bq = (const float*)d_in[4];
  const float* Wk = (const float*)d_in[5];
  const float* bk = (const float*)d_in[6];
  const float* Wv = (const float*)d_in[7];
  const float* bv = (const float*)d_in[8];
  float* out = (float*)d_out;
  char* ws = (char*)d_ws;

  const size_t nqkv = (size_t)Bn * Sn * Dn;   // elements
  unsigned short* qp   = (unsigned short*)ws;                 ws += nqkv * 2;
  unsigned short* kp   = (unsigned short*)ws;                 ws += nqkv * 2;
  unsigned short* vT   = (unsigned short*)ws;                 ws += nqkv * 2;   // [B,512,4096]
  unsigned short* qlm  = (unsigned short*)ws;                 ws += (size_t)Bn * Mn * Dn * 2;
  unsigned short* klm  = (unsigned short*)ws;                 ws += (size_t)Bn * Mn * Dn * 2;
  unsigned short* K1   = (unsigned short*)ws;                 ws += (size_t)Bn * Sn * Mn * 2;
  unsigned short* S3b  = (unsigned short*)ws;                 ws += (size_t)Bn * Mn * Sn * 2;
  unsigned short* k3vT = (unsigned short*)ws;                 ws += (size_t)Bn * Dn * Mn * 2;
  unsigned short* tmpb = (unsigned short*)ws;                 ws += (size_t)Bn * Sn * Mn * 2;
  float* S3  = (float*)ws;                                    ws += (size_t)Bn * Mn * Sn * 4;
  float* k2i = (float*)ws;                                    ws += (size_t)Bn * Mn * Mn * 4;

  const float scaling = 0.21022410381342863f;   // 512^(-1/4)
  dim3 blk(256);
  proj_kernel<<<dim3(128, 4), blk, 0, stream>>>(query, Wq, bq, qp, scaling, 0);
  proj_kernel<<<dim3(128, 4), blk, 0, stream>>>(key,   Wk, bk, kp, scaling, 0);
  proj_kernel<<<dim3(128, 4), blk, 0, stream>>>(value, Wv, bv, vT, 1.0f, 1);
  pool_kernel<<<dim3(256), blk, 0, stream>>>(qp, qlm);
  pool_kernel<<<dim3(256), blk, 0, stream>>>(kp, klm);
  k1_kernel<<<dim3(32, 4), blk, 0, stream>>>(qp, klm, K1);
  k2pinv_kernel<<<dim3(4), blk, 0, stream>>>(qlm, klm, k2i);
  k3_kernel<<<dim3(32, 4), blk, 0, stream>>>(qlm, kp, S3);
  softmax_row_kernel<<<dim3(256), blk, 0, stream>>>(S3, S3b);
  k3v_kernel<<<dim3(4, 4), blk, 0, stream>>>(S3b, vT, k3vT);
  tmp_kernel<<<dim3(32, 4), blk, 0, stream>>>(K1, k2i, tmpb);
  out_kernel<<<dim3(32, 4, 4), blk, 0, stream>>>(tmpb, k3vT, out);
}
